// Dconv_cshuffle_21827023798978
// MI455X (gfx1250) — matrix operand order repaired and checked by reading, not compiled, not run on hardware
//
#include <hip/hip_runtime.h>
#include <hip/hip_bf16.h>

typedef __attribute__((ext_vector_type(16))) __bf16 v16bf;
typedef __attribute__((ext_vector_type(8)))  __bf16 v8bf;
typedef __attribute__((ext_vector_type(8)))  float  v8f;
typedef __attribute__((ext_vector_type(4)))  float  v4f;

#define Bn   32
#define Cn   256
#define Hn   56
#define Wn   56
#define HWn  (Hn * Wn)        // 3136
#define OCn  256
#define Mtot (Bn * HWn)       // 100352
#define KBn  (Cn / 32)        // 8 k-blocks of 32 per tap
#define NBn  (OCn / 16)       // 16 n-blocks of 16

// ---------------------------------------------------------------------------
// Pass 0: weight transform.  w is OIHW fp32 (OC,C,3,3).  Emit bf16 hi/lo
// planes pre-swizzled into WMMA B-fragment order:
//   WB[t][kb][nb][lane][j], j<8 -> K=8*(lane>>4)+j, j>=8 -> K=16+8*(lane>>4)+j-8,
//   n = lane&15.  9*8*16*32*16 = 589824 elems = 2304 blocks * 256.
// ---------------------------------------------------------------------------
__global__ void wxform(const float* __restrict__ w,
                       __bf16* __restrict__ wb_hi,
                       __bf16* __restrict__ wb_lo)
{
    int idx  = blockIdx.x * blockDim.x + threadIdx.x;
    int j    = idx & 15;
    int lane = (idx >> 4)  & 31;
    int nb   = (idx >> 9)  & 15;
    int kb   = (idx >> 13) & 7;
    int t    = idx >> 16;                 // 0..8 exactly
    int n    = lane & 15;
    int k    = ((lane >> 4) << 3) + (j & 7) + ((j >> 3) << 4);    // 0..31
    int oc   = nb * 16 + n;
    int ic   = kb * 32 + k;
    int kh   = t / 3, kw = t % 3;
    float v  = w[((oc * Cn + ic) * 3 + kh) * 3 + kw];
    __bf16 hi = (__bf16)v;
    __bf16 lo = (__bf16)(v - (float)hi);
    wb_hi[idx] = hi;
    wb_lo[idx] = lo;
}

// ---------------------------------------------------------------------------
// Pass 1: fused channel shuffle + fp32 -> (hi,lo) bf16 split.
// A layout: row-major (m = b*HW + hw, c), plus ONE trailing zero row at
// m == Mtot used by the GEMM to realize conv zero-padding with
// unconditional (never-divergent) loads.  Launched with one extra block.
// ---------------------------------------------------------------------------
__global__ void shuffle_cvt(const float* __restrict__ x,
                            const int*   __restrict__ perm,
                            __bf16* __restrict__ a_hi,
                            __bf16* __restrict__ a_lo)
{
    unsigned idx = blockIdx.x * blockDim.x + threadIdx.x;   // over (Mtot+1)*Cn
    int m  = idx >> 8;
    if (m >= Mtot) {                       // the padding row
        a_hi[idx] = (__bf16)0.0f;
        a_lo[idx] = (__bf16)0.0f;
        return;
    }
    int c  = idx & (Cn - 1);
    int b  = m / HWn;
    int hw = m - b * HWn;
    int pc = perm[hw * Cn + c];
    float v = x[(b * Cn + pc) * HWn + hw];
    __bf16 hi = (__bf16)v;
    __bf16 lo = (__bf16)(v - (float)hi);
    a_hi[idx] = hi;
    a_lo[idx] = lo;
}

// ---------------------------------------------------------------------------
// Pass 2: implicit-GEMM conv.  WG = 256 threads = 8 wave32, tile 128M x 64N.
// Wave grid 4(M) x 2(N); each wave owns 32x32 via 2x2 accum tiles (v8f).
// K loop: 9 taps x 8 kblocks; per iter 12 WMMAs (2m x 2n x {hh,hl,lh}).
// Padding handled by clamping OOB rows to the zero row -> no divergence,
// no EXEC save/restore, no register re-zeroing in the mainloop.
// ---------------------------------------------------------------------------
union UA { v16bf v; v8bf h[2]; };
union UACC { v8f v; v4f q[2]; };

__global__ __launch_bounds__(256) void conv_gemm(
    const __bf16* __restrict__ a_hi, const __bf16* __restrict__ a_lo,
    const __bf16* __restrict__ wb_hi, const __bf16* __restrict__ wb_lo,
    float* __restrict__ out)
{
    const int lane   = threadIdx.x & 31;
    const int waveId = threadIdx.x >> 5;
    const int wm = waveId & 3;                     // M wave index 0..3
    const int wn = waveId >> 2;                    // N wave index 0..1
    const int n0 = blockIdx.x * 64 + wn * 32;      // oc base of wave
    const int m0 = blockIdx.y * 128 + wm * 32;     // m  base of wave
    const int lrow = lane & 15;
    const int ka0  = (lane & 16) ? 8 : 0;          // A-frag first K chunk
    const int ZROW = Mtot * Cn;                    // zero-padding row

    // per-lane row decode (tap-invariant: hoisted out of the K loop)
    int mR[2], hR[2], wRr[2];
#pragma unroll
    for (int mi = 0; mi < 2; ++mi) {
        int m  = m0 + mi * 16 + lrow;
        int b  = m / HWn;
        int hw = m - b * HWn;
        int h  = hw / Wn;
        mR[mi] = m; hR[mi] = h; wRr[mi] = hw - h * Wn;
    }

    v8f zero = {};
    v8f acc[2][2];
    acc[0][0] = zero; acc[0][1] = zero; acc[1][0] = zero; acc[1][1] = zero;

    const int nbBase = blockIdx.x * 4 + wn * 2;

    for (int t = 0; t < 9; ++t) {
        const int dh = t / 3 - 1;
        const int dw = t % 3 - 1;

        // per-lane element offsets; OOB rows clamp to the zero row
        int aoff[2];
#pragma unroll
        for (int mi = 0; mi < 2; ++mi) {
            int hh = hR[mi] + dh, ww = wRr[mi] + dw;
            bool ok = ((unsigned)hh < (unsigned)Hn) &
                      ((unsigned)ww < (unsigned)Wn);
            aoff[mi] = (ok ? (mR[mi] + dh * Wn + dw) * Cn : ZROW) + ka0;
        }
        int boff[2];
#pragma unroll
        for (int ni = 0; ni < 2; ++ni)
            boff[ni] = ((t * KBn * NBn + nbBase + ni) * 32 + lane) << 4;

        for (int kb = 0; kb < KBn; ++kb) {
            // ---- A fragments: unconditional per-lane b128 pairs ----
            UA Ah[2], Al[2];
#pragma unroll
            for (int mi = 0; mi < 2; ++mi) {
                Ah[mi].h[0] = *(const v8bf*)(a_hi + aoff[mi]);
                Ah[mi].h[1] = *(const v8bf*)(a_hi + aoff[mi] + 16);
                Al[mi].h[0] = *(const v8bf*)(a_lo + aoff[mi]);
                Al[mi].h[1] = *(const v8bf*)(a_lo + aoff[mi] + 16);
                aoff[mi] += 32;                     // next k-block
            }

            // ---- B fragments: contiguous 32B per lane, coalesced ----
            UA Bh[2], Bl[2];
#pragma unroll
            for (int ni = 0; ni < 2; ++ni) {
                Bh[ni].h[0] = *(const v8bf*)(wb_hi + boff[ni]);
                Bh[ni].h[1] = *(const v8bf*)(wb_hi + boff[ni] + 8);
                Bl[ni].h[0] = *(const v8bf*)(wb_lo + boff[ni]);
                Bl[ni].h[1] = *(const v8bf*)(wb_lo + boff[ni] + 8);
                if (kb + 1 < KBn)                   // prefetch next k-block
                    __builtin_prefetch(
                        (const void*)(wb_hi + boff[ni] + NBn * 32 * 16), 0, 1);
                boff[ni] += NBn * 32 * 16;          // next k-block
            }

            // ---- 3-product bf16 emulation of fp32 GEMM ----
#pragma unroll
            for (int mi = 0; mi < 2; ++mi)
#pragma unroll
                for (int ni = 0; ni < 2; ++ni) {
                    acc[mi][ni] = __builtin_amdgcn_wmma_f32_16x16x32_bf16(
                        false, Ah[mi].v, false, Bh[ni].v, (short)0,
                        acc[mi][ni], false, false);
                    acc[mi][ni] = __builtin_amdgcn_wmma_f32_16x16x32_bf16(
                        false, Ah[mi].v, false, Bl[ni].v, (short)0,
                        acc[mi][ni], false, false);
                    acc[mi][ni] = __builtin_amdgcn_wmma_f32_16x16x32_bf16(
                        false, Al[mi].v, false, Bh[ni].v, (short)0,
                        acc[mi][ni], false, false);
                }
        }
    }

    // ---- epilogue: lane holds 8 consecutive hw for one oc -> 2x b128 ----
#pragma unroll
    for (int mi = 0; mi < 2; ++mi)
#pragma unroll
        for (int ni = 0; ni < 2; ++ni) {
            int m_sub = m0 + mi * 16;
            int oc    = n0 + ni * 16 + (lane & 15);
            int mrow  = m_sub + (lane >> 4) * 8;   // VGPR r <-> row M=r(+8)
            int b     = m_sub / HWn;               // subtile never crosses b
            int hw    = mrow - b * HWn;            // multiple of 8 -> aligned
            float* po = out + (b * OCn + oc) * HWn + hw;
            UACC ua; ua.v = acc[mi][ni];
            *(v4f*)po       = ua.q[0];
            *(v4f*)(po + 4) = ua.q[1];
        }
}

extern "C" void kernel_launch(void* const* d_in, const int* in_sizes, int n_in,
                              void* d_out, int out_size, void* d_ws, size_t ws_size,
                              hipStream_t stream)
{
    const float* x    = (const float*)d_in[0];
    const float* w    = (const float*)d_in[1];
    const int*   perm = (const int*)d_in[2];
    float* out = (float*)d_out;

    char* ws = (char*)d_ws;
    const size_t asz = (size_t)(Mtot + 1) * Cn * sizeof(__bf16);    // +zero row
    const size_t wsz = (size_t)9 * KBn * NBn * 32 * 16 * sizeof(__bf16);
    __bf16* a_hi  = (__bf16*)ws;
    __bf16* a_lo  = (__bf16*)(ws + asz);
    __bf16* wb_hi = (__bf16*)(ws + 2 * asz);
    __bf16* wb_lo = (__bf16*)(ws + 2 * asz + wsz);

    wxform<<<(9 * KBn * NBn * 32 * 16) / 256, 256, 0, stream>>>(w, wb_hi, wb_lo);
    shuffle_cvt<<<(Mtot * Cn) / 256 + 1, 256, 0, stream>>>(x, perm, a_hi, a_lo);
    conv_gemm<<<dim3(OCn / 64, Mtot / 128), 256, 0, stream>>>(
        a_hi, a_lo, wb_hi, wb_lo, out);
}
